// EGLAVarNet_39951785787598
// MI455X (gfx1250) — matrix operand order repaired and checked by reading, not compiled, not run on hardware
//
#include <hip/hip_runtime.h>
#include <hip/hip_bf16.h>

// ---------------------------------------------------------------------------
// EGLAVarNet on MI455X (gfx1250): f16 WMMA (f32 accum) for all 1x1-conv GEMMs
// AND the kv outer-product; TDM (tensor_load_to_lds) for weight staging;
// f32 scalar for groupnorm / 3x3 convs. Everything L2-resident (192MB L2).
// ---------------------------------------------------------------------------

typedef _Float16 v16h __attribute__((ext_vector_type(16)));
typedef _Float16 v8h  __attribute__((ext_vector_type(8)));
typedef float    v8f  __attribute__((ext_vector_type(8)));
typedef uint32_t v4u  __attribute__((ext_vector_type(4)));
typedef uint32_t v8u  __attribute__((ext_vector_type(8)));

#define NPIX 65536      // 256*256 pixels per batch
#define NBATCH 2
#define C 64
#define NP 128          // pixels per GEMM workgroup
#define BSTR 72         // LDS B-tile row stride in halves (16B-aligned rows, conflict pad)
#define NLAYER 4

__device__ __forceinline__ float sigmoidf(float x) {
    return __builtin_amdgcn_rcpf(1.0f + __expf(-x));   // v_rcp_f32, not full IEEE div
}

// ---- Tensor Data Mover: 1-row tile of `nelems` f16 from global -> LDS ------
// D# per ISA 8.3/8.4: group0 {count=1, lds_addr, global_addr, type=2},
// group1 {data_size=2B, tensor_dim0=nelems, tensor_dim1=1, tile=nelems x 1}.
// Issue from one wave; drain with s_wait_tensorcnt; workgroup barrier covers rest.
__device__ __forceinline__ void tdm_load_row_f16(void* lds, const void* gsrc,
                                                 uint32_t nelems) {
    uint64_t ga = (uint64_t)(uintptr_t)gsrc;
    v4u g0;
    g0[0] = 1u;                                   // count=1 valid user descriptor
    g0[1] = (uint32_t)(uintptr_t)lds;             // lds_addr (bytes)
    g0[2] = (uint32_t)ga;                         // global_addr[31:0]
    g0[3] = (uint32_t)(ga >> 32) | (2u << 30);    // global_addr[56:32] | type=2
    v8u g1;
    g1[0] = 1u << 16;                             // data_size=1 -> 2-byte elems
    g1[1] = (nelems & 0xFFFFu) << 16;             // tensor_dim0[15:0]
    g1[2] = (nelems >> 16) | (1u << 16);          // tensor_dim0[31:16] | tensor_dim1=1
    g1[3] = (nelems & 0xFFFFu) << 16;             // tile_dim0 = nelems
    g1[4] = 1u;                                   // tile_dim1=1, tile_dim2 unused
    g1[5] = nelems;                               // tensor_dim0_stride
    g1[6] = 0u;
    g1[7] = 0u;
    asm volatile("tensor_load_to_lds %0, %1" :: "s"(g0), "s"(g1) : "memory");
}

// ---- WMMA operand builders (ISA 7.12.2 layouts) ---------------------------
// A (16x32 f16, M x K): lane<16 -> row m0+lane, K {0..7} and {16..23};
//                       lane>=16 -> row m0+lane-16, K {8..15} and {24..31}.
__device__ __forceinline__ v16h load_a(const _Float16* sW, int mrow, int kstep, int lane) {
    int kA = (lane < 16) ? 0 : 8;
    const _Float16* p = sW + mrow * 64 + kstep * 32 + kA;
    v8h lo = *reinterpret_cast<const v8h*>(p);        // 16B aligned
    v8h hi = *reinterpret_cast<const v8h*>(p + 16);   // 16B aligned
    v16h r;
#pragma unroll
    for (int i = 0; i < 8; ++i) { r[i] = lo[i]; r[i + 8] = hi[i]; }
    return r;
}

// B (32x16 f16, K x N): lane<16 -> col n, K {0..7} and {16..23}; lane>=16 -> col n, K {8..15} and {24..31}.
// sB is the transposed activation tile: sB[n_local][k], row stride BSTR halves.
__device__ __forceinline__ v16h load_b(const _Float16* sB, int nl, int kstep, int lane) {
    int kB = (lane >> 4) * 8;
    const _Float16* p = sB + nl * BSTR + kstep * 32 + kB;
    v8h lo = *reinterpret_cast<const v8h*>(p);
    v8h hi = *reinterpret_cast<const v8h*>(p + 16);
    v16h r;
#pragma unroll
    for (int i = 0; i < 8; ++i) { r[i] = lo[i]; r[i + 8] = hi[i]; }
    return r;
}

// One 16(M)x16(N) f32 tile, K=64 as two v_wmma_f32_16x16x32_f16 steps.
__device__ __forceinline__ v8f wmma_tile(const _Float16* sW, const _Float16* sB,
                                         int m0, int nlb, int lane) {
    v8f acc = {};
    int mrow = m0 + (lane & 15);
    int nl   = nlb + (lane & 15);
#pragma unroll
    for (int s = 0; s < 2; ++s) {
        v16h a = load_a(sW, mrow, s, lane);
        v16h b = load_b(sB, nl, s, lane);
        acc = __builtin_amdgcn_wmma_f32_16x16x32_f16(false, a, false, b,
                                                     (short)0, acc, false, false);
    }
    return acc;
}

// ---------------------------------------------------------------------------
__global__ void zero_kernel(float* p, int n) {
    int i = blockIdx.x * 256 + threadIdx.x;
    if (i < n) p[i] = 0.0f;
}

// ---------------------------------------------------------------------------
// 3x3 conv CIN=2 -> C=64 (input stage)
__global__ __launch_bounds__(256) void conv3x3_in_kernel(
    const float* __restrict__ x, const float* __restrict__ W,
    const float* __restrict__ bias, float* __restrict__ feat) {
    int idx = blockIdx.x * 256 + threadIdx.x;                 // [b][oc][p], 2*64*65536
    int p  = idx & (NPIX - 1);
    int oc = (idx >> 16) & 63;
    int b  = idx >> 22;
    int py = p >> 8, px = p & 255;
    float acc = bias[oc];
#pragma unroll
    for (int ci = 0; ci < 2; ++ci) {
        const float* xc = x + (((size_t)b * 2 + ci) << 16);
#pragma unroll
        for (int dy = -1; dy <= 1; ++dy) {
            int yy = py + dy; if (yy < 0 || yy > 255) continue;
#pragma unroll
            for (int dx = -1; dx <= 1; ++dx) {
                int xx = px + dx; if (xx < 0 || xx > 255) continue;
                acc += W[((oc * 2 + ci) * 3 + dy + 1) * 3 + dx + 1] * xc[(yy << 8) + xx];
            }
        }
    }
    feat[idx] = acc;
}

// 3x3 conv C=64 -> 2 + input residual (output stage)
__global__ __launch_bounds__(256) void conv3x3_out_kernel(
    const float* __restrict__ feat, const float* __restrict__ W,
    const float* __restrict__ bias, const float* __restrict__ x,
    float* __restrict__ out) {
    int idx = blockIdx.x * 256 + threadIdx.x;                 // [b][oc][p], 2*2*65536
    int p  = idx & (NPIX - 1);
    int oc = (idx >> 16) & 1;
    int b  = idx >> 17;
    int py = p >> 8, px = p & 255;
    float acc = bias[oc] + x[idx];
    for (int dy = -1; dy <= 1; ++dy) {
        int yy = py + dy; if (yy < 0 || yy > 255) continue;
        for (int dx = -1; dx <= 1; ++dx) {
            int xx = px + dx; if (xx < 0 || xx > 255) continue;
            const float* fb = feat + (((size_t)b * C) << 16) + (yy << 8) + xx;
            const float* wb = W + (oc * C * 9) + (dy + 1) * 3 + (dx + 1);
            float a = 0.f;
#pragma unroll 8
            for (int c = 0; c < C; ++c) a += wb[c * 9] * fb[(size_t)c << 16];
            acc += a;
        }
    }
    out[idx] = acc;
}

// ---------------------------------------------------------------------------
// GroupNorm partial sums per (b,group). grid = 16 groups * 32 chunks.
__global__ __launch_bounds__(256) void gn_stats_kernel(
    const float* __restrict__ feat, float* __restrict__ gsum, float* __restrict__ gsq) {
    int blk = blockIdx.x;
    int bg = blk >> 5, chunk = blk & 31;
    const float* base = feat + (size_t)bg * 524288 + (size_t)chunk * 16384;
    int t = threadIdx.x;
    float s = 0.f, q = 0.f;
    for (int i = t; i < 16384; i += 256) { float v = base[i]; s += v; q += v * v; }
    __shared__ float rs[256], rq[256];
    rs[t] = s; rq[t] = q; __syncthreads();
    for (int off = 128; off; off >>= 1) {
        if (t < off) { rs[t] += rs[t + off]; rq[t] += rq[t + off]; }
        __syncthreads();
    }
    if (t == 0) { atomicAdd(&gsum[bg], rs[0]); atomicAdd(&gsq[bg], rq[0]); }
}

__global__ void gn_finalize_kernel(const float* gsum, const float* gsq,
                                   float* mu, float* rstd) {
    int i = threadIdx.x;          // 16 (b,g) pairs
    if (i < 16) {
        const float inv = 1.0f / 524288.0f;
        float m = gsum[i] * inv;
        float v = gsq[i] * inv - m * m;
        mu[i] = m;
        rstd[i] = rsqrtf(v + 1e-5f);
    }
}

// normalize + cast to f16:  xn = (feat-mu)*rstd*gamma + beta
__global__ __launch_bounds__(256) void normalize_kernel(
    const float* __restrict__ feat, const float* __restrict__ mu,
    const float* __restrict__ rstd, const float* __restrict__ gamma,
    const float* __restrict__ beta, _Float16* __restrict__ xn) {
    int idx = blockIdx.x * 256 + threadIdx.x;   // 2*64*65536
    int bc = idx >> 16;
    int b = bc >> 6, c = bc & 63;
    int bg = b * 8 + (c >> 3);
    float v = (feat[idx] - mu[bg]) * rstd[bg] * gamma[c] + beta[c];
    xn[idx] = (_Float16)v;
}

// pack Wq|Wk|Wv|Wg -> f16 [320][64], biases -> f32 [320]
__global__ void pack_w_kernel(const float* Wq, const float* Wk, const float* Wv,
                              const float* Wg, const float* bq, const float* bk,
                              const float* bv, const float* bg,
                              _Float16* Wcat, float* bcat) {
    int i = blockIdx.x * 256 + threadIdx.x;     // 320*64
    if (i < 320 * 64) {
        int r = i >> 6, cc = i & 63;
        float w = (r < 64)  ? Wq[r * 64 + cc]
                : (r < 128) ? Wk[(r - 64) * 64 + cc]
                : (r < 192) ? Wv[(r - 128) * 64 + cc]
                            : Wg[(r - 192) * 64 + cc];
        Wcat[i] = (_Float16)w;
    }
    if (i < 320) {
        bcat[i] = (i < 64)  ? bq[i]
                : (i < 128) ? bk[i - 64]
                : (i < 192) ? bv[i - 128]
                            : bg[i - 192];
    }
}

// ---------------------------------------------------------------------------
// QKVG WMMA GEMM: D[320 x N] = Wcat[320x64] * xn[64 x N], fused epilogue:
//   rows   0- 63: q_s = (.)*C^-0.5       -> qbuf  (f16)
//   rows  64-127: k   (kept, paired)
//   rows 128-191: v                      -> vbuf  (f16)
//   rows 192-255: f   = sigmoid(.)       -> fsum accumulation
//   rows 256-319: i   = sigmoid(.)       -> kg = k*i -> kgbuf (f16)
__global__ __launch_bounds__(256) void qkvg_gemm_kernel(
    const _Float16* __restrict__ xn, const _Float16* __restrict__ Wcat,
    const float* __restrict__ bcat, _Float16* __restrict__ qbuf,
    _Float16* __restrict__ kgbuf, _Float16* __restrict__ vbuf,
    float* __restrict__ fsum) {
    __shared__ __align__(16) _Float16 sW[320 * 64];
    __shared__ __align__(16) _Float16 sB[NP * BSTR];
    __shared__ float sF[64];

    int t = threadIdx.x;
    int blk = blockIdx.x;
    int b  = blk >> 9;                 // 512 pixel-blocks per batch
    int n0 = (blk & 511) * NP;

    // weights: one TDM DMA (40KB) issued by wave 0 instead of 160 VMEM ops/thread
    if (t < 32) {
        tdm_load_row_f16(sW, Wcat, 320 * 64);
        __builtin_amdgcn_s_wait_tensorcnt(0);
    }
    if (t < 64) sF[t] = 0.0f;
    const _Float16* xb = xn + ((size_t)b * C << 16);
    for (int e = t; e < C * NP; e += 256) {
        int k = e / NP, nl = e % NP;                // coalesced read, transposed store
        sB[nl * BSTR + k] = xb[((size_t)k << 16) + n0 + nl];
    }
    __syncthreads();

    int wave = t >> 5, lane = t & 31;
    const float scaleq = 0.125f;                    // C^-0.5
    int hi8 = (lane >= 16) ? 8 : 0;

    for (int u = wave; u < 128; u += 8) {           // 8 N-subtiles * 16 work units
        int nsub = u >> 4;
        int kind = u & 15;
        int nlb  = nsub * 16;
        int ncol = n0 + nlb + (lane & 15);
        if (kind < 12) {
            int m0 = (kind < 4) ? kind * 16
                   : (kind < 8) ? 128 + (kind - 4) * 16
                                : 192 + (kind - 8) * 16;
            v8f acc = wmma_tile(sW, sB, m0, nlb, lane);
            if (kind < 4) {                          // q
#pragma unroll
                for (int j = 0; j < 8; ++j) {
                    int m = m0 + j + hi8;
                    float val = (acc[j] + bcat[m]) * scaleq;
                    qbuf[(((size_t)b * C + m) << 16) + ncol] = (_Float16)val;
                }
            } else if (kind < 8) {                   // v
#pragma unroll
                for (int j = 0; j < 8; ++j) {
                    int m = m0 + j + hi8;
                    vbuf[(((size_t)b * C + (m - 128)) << 16) + ncol] =
                        (_Float16)(acc[j] + bcat[m]);
                }
            } else {                                 // f gate -> row sums
#pragma unroll
                for (int j = 0; j < 8; ++j) {
                    int m = m0 + j + hi8;
                    atomicAdd(&sF[m - 192], sigmoidf(acc[j] + bcat[m]));
                }
            }
        } else {                                     // paired k & i -> kg = k * sigmoid(i)
            int jm = kind - 12;
            int mk0 = 64 + jm * 16, mi0 = 256 + jm * 16;
            v8f ak = wmma_tile(sW, sB, mk0, nlb, lane);
            v8f ai = wmma_tile(sW, sB, mi0, nlb, lane);
#pragma unroll
            for (int j = 0; j < 8; ++j) {
                int mk = mk0 + j + hi8, mi = mi0 + j + hi8;
                float kval = ak[j] + bcat[mk];
                float ig   = sigmoidf(ai[j] + bcat[mi]);
                kgbuf[(((size_t)b * C + (mk - 64)) << 16) + ncol] =
                    (_Float16)(kval * ig);
            }
        }
    }
    __syncthreads();
    if (t < 64) atomicAdd(&fsum[b * 64 + t], sF[t]);
}

// ---------------------------------------------------------------------------
// kv outer product on WMMA: knw[c,d] += sum_n kg[c,n]*v[d,n].
// A = kg (M=c, K=n, row-major -> A lane chunks contiguous);
// B-operand lane needs v[d, nb+kB+0..7] and v[d, nb+kB+16..23] -- contiguous in [d][n] storage, so the
// operands come straight from global (L2) with 16B loads, no transpose staging.
__global__ __launch_bounds__(256) void knowledge_kernel(
    const _Float16* __restrict__ kg, const _Float16* __restrict__ v,
    float* __restrict__ knw) {
    int t = threadIdx.x, wave = t >> 5, lane = t & 31;
    int blk = blockIdx.x;
    int b  = blk >> 6;                 // 64 blocks per batch
    int n0 = (blk & 63) * 1024;        // 1024-pixel span -> 32 K-steps
    int kA  = (lane < 16) ? 0 : 8;
    int kB  = (lane >> 4) * 8;
    int hi8 = (lane >= 16) ? 8 : 0;
#pragma unroll
    for (int tile = 0; tile < 2; ++tile) {
        int tid = wave * 2 + tile;     // 16 (c0,d0) tiles
        int c0 = (tid >> 2) * 16, d0 = (tid & 3) * 16;
        const _Float16* arow = kg + (((size_t)b * C + c0 + (lane & 15)) << 16) + n0;
        const _Float16* brow = v  + (((size_t)b * C + d0 + (lane & 15)) << 16) + n0;
        v8f acc = {};
        for (int kk = 0; kk < 32; ++kk) {
            int nb = kk * 32;
            __builtin_prefetch(arow + nb + 256, 0, 3);   // global_prefetch_b8
            __builtin_prefetch(brow + nb + 256, 0, 3);
            v8h a_lo = *reinterpret_cast<const v8h*>(arow + nb + kA);
            v8h a_hi = *reinterpret_cast<const v8h*>(arow + nb + kA + 16);
            v8h b_lo = *reinterpret_cast<const v8h*>(brow + nb + kB);
            v8h b_hi = *reinterpret_cast<const v8h*>(brow + nb + kB + 16);
            v16h av, bw;
#pragma unroll
            for (int i = 0; i < 8; ++i) {
                av[i] = a_lo[i]; av[i + 8] = a_hi[i];
                bw[i] = b_lo[i]; bw[i + 8] = b_hi[i];
            }
            acc = __builtin_amdgcn_wmma_f32_16x16x32_f16(false, av, false, bw,
                                                         (short)0, acc, false, false);
        }
#pragma unroll
        for (int j = 0; j < 8; ++j) {
            int c = c0 + j + hi8;
            int d = d0 + (lane & 15);
            atomicAdd(&knw[(((size_t)b * C + c) * C) + d], acc[j]);
        }
    }
}

// mem = (fsum/N) * prev_mem + knw/N   -> written straight into d_out hidden slab
__global__ void mem_update_kernel(const float* __restrict__ fsum,
                                  const float* __restrict__ knw,
                                  const float* __restrict__ hidden_in,
                                  float* __restrict__ hidden_out, int lyr) {
    int i = blockIdx.x * 256 + threadIdx.x;      // [b][c][d], 2*64*64
    int b = i >> 12, c = (i >> 6) & 63, d = i & 63;
    const float invN = 1.0f / 65536.0f;
    float fm = fsum[b * 64 + c] * invN;
    size_t hi = ((((size_t)b * NLAYER + lyr) * C) + c) * C + d;
    hidden_out[hi] = fm * hidden_in[hi] + knw[i] * invN;
}

// Weff[b][o][c] = sum_d Wp[o][d] * mem[b][c][d]   (f16)  -- fuses retrieval+proj
__global__ void weff_kernel(const float* __restrict__ mem, const float* __restrict__ Wp,
                            _Float16* __restrict__ Weff, int lyr) {
    int i = blockIdx.x * 256 + threadIdx.x;      // [b][o][c], 2*64*64
    int b = i >> 12, o = (i >> 6) & 63, cc = i & 63;
    const float* m = mem + ((((size_t)b * NLAYER + lyr) * C) + cc) * C;
    float acc = 0.0f;
#pragma unroll 16
    for (int d = 0; d < C; ++d) acc += Wp[o * C + d] * m[d];
    Weff[i] = (_Float16)acc;
}

// ---------------------------------------------------------------------------
// PROJ WMMA GEMM: feat[o,n] = sum_c Weff[b][o][c]*q_s[c,n] + bp[o] + feat[o,n]
__global__ __launch_bounds__(256) void proj_gemm_kernel(
    const _Float16* __restrict__ qs, const _Float16* __restrict__ Weff,
    const float* __restrict__ bp, float* __restrict__ feat) {
    __shared__ __align__(16) _Float16 sW[64 * 64];
    __shared__ __align__(16) _Float16 sB[NP * BSTR];
    int t = threadIdx.x;
    int blk = blockIdx.x;
    int b  = blk >> 9;
    int n0 = (blk & 511) * NP;

    if (t < 32) {                                   // 8KB weight tile via TDM
        tdm_load_row_f16(sW, Weff + (size_t)b * 64 * 64, 64 * 64);
        __builtin_amdgcn_s_wait_tensorcnt(0);
    }
    const _Float16* qb = qs + ((size_t)b * C << 16);
    for (int e = t; e < C * NP; e += 256) {
        int k = e / NP, nl = e % NP;
        sB[nl * BSTR + k] = qb[((size_t)k << 16) + n0 + nl];
    }
    __syncthreads();

    int wave = t >> 5, lane = t & 31;
    int hi8 = (lane >= 16) ? 8 : 0;
    for (int u = wave; u < 32; u += 8) {         // 8 N-subtiles * 4 M-tiles
        int nsub = u >> 2;
        int m0   = (u & 3) * 16;
        int nlb  = nsub * 16;
        int ncol = n0 + nlb + (lane & 15);
        v8f acc = wmma_tile(sW, sB, m0, nlb, lane);
#pragma unroll
        for (int j = 0; j < 8; ++j) {
            int m = m0 + j + hi8;
            size_t o = (((size_t)b * C + m) << 16) + ncol;
            feat[o] = acc[j] + bp[m] + feat[o];   // residual
        }
    }
}

// ---------------------------------------------------------------------------
extern "C" void kernel_launch(void* const* d_in, const int* in_sizes, int n_in,
                              void* d_out, int out_size, void* d_ws, size_t ws_size,
                              hipStream_t stream) {
    (void)in_sizes; (void)n_in; (void)out_size; (void)ws_size;
    // inputs (setup_inputs order)
    const float* x      = (const float*)d_in[0];   // [2,2,256,256]
    const float* hidden = (const float*)d_in[1];   // [2,4,64,64]
    const float* W_in   = (const float*)d_in[2];   // [64,2,3,3]
    const float* b_in   = (const float*)d_in[3];   // [64]
    const float* gamma  = (const float*)d_in[4];   // [4,64]
    const float* beta   = (const float*)d_in[5];   // [4,64]
    const float* Wq     = (const float*)d_in[6];   // [4,64,64]
    const float* bq     = (const float*)d_in[7];   // [4,64]
    const float* Wk     = (const float*)d_in[8];
    const float* bk     = (const float*)d_in[9];
    const float* Wv     = (const float*)d_in[10];
    const float* bv     = (const float*)d_in[11];
    const float* Wg     = (const float*)d_in[12];  // [4,128,64]
    const float* bg     = (const float*)d_in[13];  // [4,128]
    const float* Wp     = (const float*)d_in[14];  // [4,64,64]
    const float* bp     = (const float*)d_in[15];  // [4,64]
    const float* W_out  = (const float*)d_in[16];  // [2,64,3,3]
    const float* b_out  = (const float*)d_in[17];  // [2]

    float* out_img    = (float*)d_out;             // [2,2,256,256] = 262144 f32
    float* hidden_out = (float*)d_out + 2 * 2 * 256 * 256;  // [2,4,64,64]

    // workspace layout (256B aligned slabs)
    char* ws = (char*)d_ws;
    constexpr size_t OFF_FEAT  = 0;                               // f32 [2][64][65536]
    constexpr size_t OFF_XN    = OFF_FEAT  + 8388608ull * 4;      // f16 same shape
    constexpr size_t OFF_Q     = OFF_XN    + 8388608ull * 2;
    constexpr size_t OFF_KG    = OFF_Q     + 8388608ull * 2;
    constexpr size_t OFF_V     = OFF_KG    + 8388608ull * 2;
    constexpr size_t OFF_WCAT  = OFF_V     + 8388608ull * 2;      // f16 [320][64]
    constexpr size_t OFF_BCAT  = OFF_WCAT  + 40960 + 1024;        // f32 [320]
    constexpr size_t OFF_WEFF  = OFF_BCAT  + 1280 + 768;          // f16 [2][64][64]
    constexpr size_t OFF_STATS = OFF_WEFF  + 16384;               // gsum16|gsq16|mu16|rstd16
    constexpr size_t OFF_FSUM  = OFF_STATS + 256;                 // f32 [2][64]
    constexpr size_t OFF_KNW   = OFF_FSUM  + 512;                 // f32 [2][64][64]

    float*    feat  = (float*)(ws + OFF_FEAT);
    _Float16* xn    = (_Float16*)(ws + OFF_XN);
    _Float16* qbuf  = (_Float16*)(ws + OFF_Q);
    _Float16* kgbuf = (_Float16*)(ws + OFF_KG);
    _Float16* vbuf  = (_Float16*)(ws + OFF_V);
    _Float16* Wcat  = (_Float16*)(ws + OFF_WCAT);
    float*    bcat  = (float*)(ws + OFF_BCAT);
    _Float16* Weff  = (_Float16*)(ws + OFF_WEFF);
    float*    gsum  = (float*)(ws + OFF_STATS);
    float*    gsq   = gsum + 16;
    float*    mu    = gsum + 32;
    float*    rstd  = gsum + 48;
    float*    fsum  = (float*)(ws + OFF_FSUM);
    float*    knw   = (float*)(ws + OFF_KNW);

    // input 3x3 conv
    conv3x3_in_kernel<<<8388608 / 256, 256, 0, stream>>>(x, W_in, b_in, feat);

    for (int lyr = 0; lyr < NLAYER; ++lyr) {
        // zero all layer accumulators (gsum..knw contiguous region)
        int nz = (int)((OFF_KNW + 32768 - OFF_STATS) / 4);
        zero_kernel<<<(nz + 255) / 256, 256, 0, stream>>>(gsum, nz);

        gn_stats_kernel<<<512, 256, 0, stream>>>(feat, gsum, gsq);
        gn_finalize_kernel<<<1, 16, 0, stream>>>(gsum, gsq, mu, rstd);

        pack_w_kernel<<<80, 256, 0, stream>>>(
            Wq + lyr * 4096, Wk + lyr * 4096, Wv + lyr * 4096, Wg + lyr * 8192,
            bq + lyr * 64, bk + lyr * 64, bv + lyr * 64, bg + lyr * 128,
            Wcat, bcat);

        normalize_kernel<<<8388608 / 256, 256, 0, stream>>>(
            feat, mu, rstd, gamma + lyr * 64, beta + lyr * 64, xn);

        qkvg_gemm_kernel<<<1024, 256, 0, stream>>>(
            xn, Wcat, bcat, qbuf, kgbuf, vbuf, fsum);

        knowledge_kernel<<<128, 256, 0, stream>>>(kgbuf, vbuf, knw);

        mem_update_kernel<<<32, 256, 0, stream>>>(fsum, knw, hidden, hidden_out, lyr);
        weff_kernel<<<32, 256, 0, stream>>>(hidden_out, Wp + lyr * 4096, Weff, lyr);

        proj_gemm_kernel<<<1024, 256, 0, stream>>>(qbuf, Weff, bp + lyr * 64, feat);
    }

    conv3x3_out_kernel<<<262144 / 256, 256, 0, stream>>>(feat, W_out, b_out, x, out_img);
}
